// MultiheadAttention_84748294685191
// MI455X (gfx1250) — matrix operand order repaired and checked by reading, not compiled, not run on hardware
//
#include <hip/hip_runtime.h>

typedef unsigned short u16;
typedef unsigned int   u32;
typedef __attribute__((ext_vector_type(16))) __bf16 v16bf;
typedef __attribute__((ext_vector_type(8)))  float  v8f;
typedef __attribute__((ext_vector_type(4)))  u32    u32x4;
typedef __attribute__((ext_vector_type(8)))  int    i32x8;
typedef __attribute__((ext_vector_type(4)))  int    i32x4;

union Frag16 { v16bf v; uint4 q[2]; u16 s[16]; };
union U4     { uint4 q; u16 s[8]; };

#if defined(__has_builtin)
#if __has_builtin(__builtin_amdgcn_tensor_load_to_lds) && __has_builtin(__builtin_amdgcn_s_wait_tensorcnt)
#define USE_TDM 1
#endif
#endif
#ifndef USE_TDM
#define USE_TDM 0
#endif

__device__ __forceinline__ u16 f2bf(float f) {
  u32 u = __float_as_uint(f);
  u += 0x7FFFu + ((u >> 16) & 1u);     // round-to-nearest-even
  return (u16)(u >> 16);
}

#define WMMA_BF16(A, B, C) \
  __builtin_amdgcn_wmma_f32_16x16x32_bf16(false, (A), false, (B), (short)0, (C), false, false)

// ---------------------------------------------------------------------------
// f32 -> bf16 conversion, 8 elements / thread (b128 in, b128 out)
// ---------------------------------------------------------------------------
__global__ __launch_bounds__(256)
void convert_f32_bf16(const float* __restrict__ src, u16* __restrict__ dst, int n) {
  int i = (blockIdx.x * blockDim.x + threadIdx.x) * 8;
  if (i >= n) return;
  float4 a = *(const float4*)(src + i);
  float4 b = *(const float4*)(src + i + 4);
  U4 o;
  o.s[0] = f2bf(a.x); o.s[1] = f2bf(a.y); o.s[2] = f2bf(a.z); o.s[3] = f2bf(a.w);
  o.s[4] = f2bf(b.x); o.s[5] = f2bf(b.y); o.s[6] = f2bf(b.z); o.s[7] = f2bf(b.w);
  *(uint4*)(dst + i) = o.q;
}

// ---------------------------------------------------------------------------
// Tiled bf16 WMMA GEMM: C[M x N] = A[M x K] * B[K x N] (+ bias)
// Block: 256 threads (8 waves), tile 128x128, BK=32.
// Wave w: rows (w&3)*32..+31, cols (w>>2)*64..+63  -> 2x4 accumulators,
// 8 WMMA per K-step. A tile staged into LDS by the Tensor Data Mover
// (wave 0 issues tensor_load_to_lds, waits TENSORcnt, then block barrier);
// B tile loaded b128 + transposed into [n][k] LDS so B fragments are
// contiguous ds_load_b128 pairs.
// ---------------------------------------------------------------------------
template<bool HAS_BIAS, bool OUT_F32>
__global__ __launch_bounds__(256)
void gemm_bf16_wmma(const u16* __restrict__ A, const u16* __restrict__ Bm,
                    const float* __restrict__ bias, void* __restrict__ Cout,
                    int M, int N, int K) {
  __shared__ __align__(16) u16 As[128][32];   // [m][k], pitch 64B
  __shared__ __align__(16) u16 Bs[128][40];   // [n][k] transposed, pitch 80B

  const int tid  = threadIdx.x;
  const int lane = tid & 31;
  const int wid  = tid >> 5;
  const int wrow = wid & 3;        // 0..3  (32-row slab)
  const int wcol = wid >> 2;       // 0..1  (64-col slab)
  const int lm   = lane & 15;
  const int hi   = lane >> 4;
  const int rowBase = blockIdx.y * 128;
  const int colBase = blockIdx.x * 128;

  // B cooperative-load indices: 32(k) x 128(n) tile, 16 u16 per thread
  const int bk = tid >> 3;         // 0..31
  const int bn = (tid & 7) * 16;   // 0..112

#if !USE_TDM
  // fallback A cooperative-load indices: 128 x 32 tile, 16 u16 per thread
  const int ar = tid >> 1;         // 0..127
  const int ac = (tid & 1) * 16;   // 0 or 16
#endif

  v8f acc[2][4] = {};

#if USE_TDM
  const unsigned ldsA = (unsigned)(unsigned long long)(&As[0][0]);
#endif

  for (int kk = 0; kk < K; kk += 32) {
    __syncthreads();

#if USE_TDM
    if (wid == 0) {
      // D# group 0: count=1 | lds_addr | global_addr(57b) | type=2
      unsigned long long ga =
          (unsigned long long)(const void*)(A + (size_t)rowBase * K + kk);
      u32x4 g0;
      g0.x = 1u;
      g0.y = ldsA;
      g0.z = (u32)ga;
      g0.w = ((u32)(ga >> 32) & 0x1FFFFFFu) | (2u << 30);
      // D# group 1: data_size=2B; tensor_dim0=K, tensor_dim1=M;
      // tile_dim0=32 (K-dir, contiguous), tile_dim1=128 rows; stride0=K.
      i32x8 g1;
      g1[0] = 0x10000;                                   // data_size code 1
      g1[1] = (int)((K & 0xFFFF) << 16);                 // tensor_dim0[15:0]
      g1[2] = (int)(((K >> 16) & 0xFFFF) | ((M & 0xFFFF) << 16));
      g1[3] = (int)(((M >> 16) & 0xFFFF) | (32u << 16)); // tile_dim0=32
      g1[4] = 128;                                       // tile_dim1=128
      g1[5] = K;                                         // stride0[31:0]
      g1[6] = 0;
      g1[7] = 0;
      i32x4 gz = {0, 0, 0, 0};
#if __clang_major__ >= 23
      i32x8 gz8 = {0, 0, 0, 0, 0, 0, 0, 0};
      __builtin_amdgcn_tensor_load_to_lds(g0, g1, gz, gz, gz8, 0);
#else
      __builtin_amdgcn_tensor_load_to_lds(g0, g1, gz, gz, 0);
#endif
    }
#else
    {
      const u16* ap = A + (size_t)(rowBase + ar) * K + kk + ac;
      uint4 a0 = *(const uint4*)ap;
      uint4 a1 = *(const uint4*)(ap + 8);
      *(uint4*)&As[ar][ac]     = a0;
      *(uint4*)&As[ar][ac + 8] = a1;
    }
#endif

    // B tile: b128 loads, transpose-scatter into Bs[n][k]
    {
      const u16* bp = Bm + (size_t)(kk + bk) * N + colBase + bn;
      U4 b0, b1;
      b0.q = *(const uint4*)bp;
      b1.q = *(const uint4*)(bp + 8);
      #pragma unroll
      for (int e = 0; e < 8; ++e) {
        Bs[bn + e][bk]     = b0.s[e];
        Bs[bn + 8 + e][bk] = b1.s[e];
      }
    }

#if USE_TDM
    if (wid == 0) __builtin_amdgcn_s_wait_tensorcnt(0);
#endif
    __syncthreads();

    // A fragments (two 16-row m-tiles)
    Frag16 af0, af1;
    const int m0  = wrow * 32 + lm;
    const int ak0 = hi * 8;
    af0.q[0] = *(const uint4*)&As[m0][ak0];
    af0.q[1] = *(const uint4*)&As[m0][ak0 + 16];
    af1.q[0] = *(const uint4*)&As[m0 + 16][ak0];
    af1.q[1] = *(const uint4*)&As[m0 + 16][ak0 + 16];

    // B fragments (four 16-col n-tiles)
    Frag16 bf[4];
    const int bk0 = hi * 8;
    #pragma unroll
    for (int nt = 0; nt < 4; ++nt) {
      const int n = wcol * 64 + nt * 16 + lm;
      bf[nt].q[0] = *(const uint4*)&Bs[n][bk0];
      bf[nt].q[1] = *(const uint4*)&Bs[n][bk0 + 16];
    }

    #pragma unroll
    for (int nt = 0; nt < 4; ++nt) {
      acc[0][nt] = WMMA_BF16(af0.v, bf[nt].v, acc[0][nt]);
      acc[1][nt] = WMMA_BF16(af1.v, bf[nt].v, acc[1][nt]);
    }
  }

  // Epilogue; C layout: VGPR r -> row r (lanes 0-15) / row r+8 (lanes 16-31)
  #pragma unroll
  for (int mt = 0; mt < 2; ++mt) {
    const int rowW = rowBase + wrow * 32 + mt * 16 + hi * 8;
    #pragma unroll
    for (int nt = 0; nt < 4; ++nt) {
      const int col = colBase + wcol * 64 + nt * 16 + lm;
      const float bb = HAS_BIAS ? bias[col] : 0.f;
      #pragma unroll
      for (int r = 0; r < 8; ++r) {
        float v = acc[mt][nt][r] + bb;
        if (OUT_F32) ((float*)Cout)[(size_t)(rowW + r) * N + col] = v;
        else         ((u16*)Cout)[(size_t)(rowW + r) * N + col]   = f2bf(v);
      }
    }
  }
}

// ---------------------------------------------------------------------------
// Flash attention with relative-position bias.
// Grid: (S/128, H, B). Block: 256 threads = 8 waves, each wave = 16 query rows.
// logits = scores*(1/sqrt(64) + 64^-0.25) + rel_bias*64^-0.25
// ---------------------------------------------------------------------------
__global__ __launch_bounds__(256)
void flash_attn_relbias(const u16* __restrict__ Q, const u16* __restrict__ Kg,
                        const u16* __restrict__ Vg, const float* __restrict__ pos_bias,
                        u16* __restrict__ AO) {
  __shared__ __align__(16) u16 Ps[8][16][40];   // per-wave P (16x32) staging
  __shared__ __align__(16) u16 Vt[64][40];      // V tile transposed [hd][key]

  const int tid  = threadIdx.x;
  const int lane = tid & 31;
  const int wid  = tid >> 5;
  const int lm   = lane & 15;
  const int hi   = lane >> 4;

  const int h   = blockIdx.y;
  const int b   = blockIdx.z;
  const int qb0 = blockIdx.x * 128 + wid * 16;
  const size_t tokBase = (size_t)b * 2048;
  const int colH = h * 64;

  // Q fragments (hd 0..31 and 32..63), loaded once from global
  Frag16 qf0, qf1;
  {
    const size_t rowOff = (tokBase + qb0 + lm) * 1024 + colH;
    const int k0 = hi * 8;
    qf0.q[0] = *(const uint4*)(Q + rowOff + k0);
    qf0.q[1] = *(const uint4*)(Q + rowOff + k0 + 16);
    qf1.q[0] = *(const uint4*)(Q + rowOff + 32 + k0);
    qf1.q[1] = *(const uint4*)(Q + rowOff + 32 + k0 + 16);
  }

  v8f o0 = {}, o1 = {}, o2 = {}, o3 = {};
  float mrow[8], lrow[8];
  #pragma unroll
  for (int r = 0; r < 8; ++r) { mrow[r] = -1e30f; lrow[r] = 0.f; }

  const float CS = 0.125f + 0.35355339f;   // 1/sqrt(64) + 64^-0.25
  const float CB = 0.35355339f;            // 64^-0.25

  const int vkey = tid >> 3;       // 0..31
  const int vn8  = (tid & 7) * 8;  // 0..56

  for (int kb = 0; kb < 2048; kb += 32) {
    __syncthreads();
    // cooperative V tile (32 keys x 64 hd) -> transposed LDS [hd][key]
    U4 vv;
    vv.q = *(const uint4*)(Vg + (tokBase + kb + vkey) * 1024 + colH + vn8);
    #pragma unroll
    for (int e = 0; e < 8; ++e) Vt[vn8 + e][vkey] = vv.s[e];
    __syncthreads();

    // K fragments straight from global: K's [key][hd] layout IS B-frag layout
    Frag16 kf00, kf01, kf10, kf11;   // [hd-chunk][n-tile]
    {
      const int kk0 = hi * 8;
      const size_t r0 = (tokBase + kb + lm) * 1024 + colH;
      const size_t r1 = (tokBase + kb + 16 + lm) * 1024 + colH;
      kf00.q[0] = *(const uint4*)(Kg + r0 + kk0);
      kf00.q[1] = *(const uint4*)(Kg + r0 + kk0 + 16);
      kf10.q[0] = *(const uint4*)(Kg + r0 + 32 + kk0);
      kf10.q[1] = *(const uint4*)(Kg + r0 + 32 + kk0 + 16);
      kf01.q[0] = *(const uint4*)(Kg + r1 + kk0);
      kf01.q[1] = *(const uint4*)(Kg + r1 + kk0 + 16);
      kf11.q[0] = *(const uint4*)(Kg + r1 + 32 + kk0);
      kf11.q[1] = *(const uint4*)(Kg + r1 + 32 + kk0 + 16);
    }

    v8f s0 = {}, s1 = {};
    s0 = WMMA_BF16(qf0.v, kf00.v, s0);
    s0 = WMMA_BF16(qf1.v, kf10.v, s0);
    s1 = WMMA_BF16(qf0.v, kf01.v, s1);
    s1 = WMMA_BF16(qf1.v, kf11.v, s1);

    // logits + rel bias, online softmax (row stats per VGPR slot, 16-lane halves)
    float p0[8], p1[8], alpha[8];
    #pragma unroll
    for (int r = 0; r < 8; ++r) {
      int i  = qb0 + r + hi * 8;
      int j0 = kb + lm;
      int d0 = i - j0;        d0 = d0 < -1023 ? -1023 : (d0 > 1023 ? 1023 : d0);
      int d1 = i - j0 - 16;   d1 = d1 < -1023 ? -1023 : (d1 > 1023 ? 1023 : d1);
      float l0 = s0[r] * CS + pos_bias[(d0 + 1023) * 16 + h] * CB;
      float l1 = s1[r] * CS + pos_bias[(d1 + 1023) * 16 + h] * CB;
      float rm = fmaxf(l0, l1);
      rm = fmaxf(rm, __shfl_xor(rm, 1, 32));
      rm = fmaxf(rm, __shfl_xor(rm, 2, 32));
      rm = fmaxf(rm, __shfl_xor(rm, 4, 32));
      rm = fmaxf(rm, __shfl_xor(rm, 8, 32));
      float mn = fmaxf(mrow[r], rm);
      alpha[r] = __expf(mrow[r] - mn);
      mrow[r]  = mn;
      float e0 = __expf(l0 - mn);
      float e1 = __expf(l1 - mn);
      p0[r] = e0; p1[r] = e1;
      float rs = e0 + e1;
      rs += __shfl_xor(rs, 1, 32);
      rs += __shfl_xor(rs, 2, 32);
      rs += __shfl_xor(rs, 4, 32);
      rs += __shfl_xor(rs, 8, 32);
      lrow[r] = lrow[r] * alpha[r] + rs;
    }

    #pragma unroll
    for (int r = 0; r < 8; ++r) {
      o0[r] *= alpha[r]; o1[r] *= alpha[r]; o2[r] *= alpha[r]; o3[r] *= alpha[r];
    }

    // P: accumulator layout -> A-fragment layout via per-wave LDS round trip
    #pragma unroll
    for (int r = 0; r < 8; ++r) {
      int prow = r + hi * 8;
      Ps[wid][prow][lm]      = f2bf(p0[r]);
      Ps[wid][prow][lm + 16] = f2bf(p1[r]);
    }
    Frag16 pf;
    {
      const int k0 = hi * 8;
      pf.q[0] = *(const uint4*)&Ps[wid][lm][k0];
      pf.q[1] = *(const uint4*)&Ps[wid][lm][k0 + 16];
    }

    // V B-fragments from transposed LDS tile
    Frag16 vf0, vf1, vf2, vf3;
    {
      const int k0 = hi * 8;
      vf0.q[0] = *(const uint4*)&Vt[lm][k0];       vf0.q[1] = *(const uint4*)&Vt[lm][k0 + 16];
      vf1.q[0] = *(const uint4*)&Vt[lm + 16][k0];  vf1.q[1] = *(const uint4*)&Vt[lm + 16][k0 + 16];
      vf2.q[0] = *(const uint4*)&Vt[lm + 32][k0];  vf2.q[1] = *(const uint4*)&Vt[lm + 32][k0 + 16];
      vf3.q[0] = *(const uint4*)&Vt[lm + 48][k0];  vf3.q[1] = *(const uint4*)&Vt[lm + 48][k0 + 16];
    }

    o0 = WMMA_BF16(pf.v, vf0.v, o0);
    o1 = WMMA_BF16(pf.v, vf1.v, o1);
    o2 = WMMA_BF16(pf.v, vf2.v, o2);
    o3 = WMMA_BF16(pf.v, vf3.v, o3);
  }

  // normalize and store bf16 [token][1024]
  #pragma unroll
  for (int r = 0; r < 8; ++r) {
    float inv = 1.0f / lrow[r];
    size_t rowOff = (tokBase + qb0 + r + hi * 8) * 1024 + colH + lm;
    AO[rowOff]      = f2bf(o0[r] * inv);
    AO[rowOff + 16] = f2bf(o1[r] * inv);
    AO[rowOff + 32] = f2bf(o2[r] * inv);
    AO[rowOff + 48] = f2bf(o3[r] * inv);
  }
}

// ---------------------------------------------------------------------------
extern "C" void kernel_launch(void* const* d_in, const int* in_sizes, int n_in,
                              void* d_out, int out_size, void* d_ws, size_t ws_size,
                              hipStream_t stream) {
  (void)in_sizes; (void)n_in; (void)out_size; (void)ws_size;
  const float* x  = (const float*)d_in[0];
  const float* Wq = (const float*)d_in[1];
  const float* bq = (const float*)d_in[2];
  const float* Wk = (const float*)d_in[3];
  const float* Wv = (const float*)d_in[4];
  const float* bv = (const float*)d_in[5];
  const float* Wo = (const float*)d_in[6];
  const float* bo = (const float*)d_in[7];
  const float* pos_bias = (const float*)d_in[8];

  const int M = 4096, D = 1024;   // M = B*S
  char* ws = (char*)d_ws;
  u16* xb  = (u16*)ws; ws += (size_t)M * D * 2;
  u16* Wqb = (u16*)ws; ws += (size_t)D * D * 2;
  u16* Wkb = (u16*)ws; ws += (size_t)D * D * 2;
  u16* Wvb = (u16*)ws; ws += (size_t)D * D * 2;
  u16* Wob = (u16*)ws; ws += (size_t)D * D * 2;
  u16* Qb  = (u16*)ws; ws += (size_t)M * D * 2;
  u16* Kb  = (u16*)ws; ws += (size_t)M * D * 2;
  u16* Vb  = (u16*)ws; ws += (size_t)M * D * 2;
  u16* AOb = (u16*)ws; ws += (size_t)M * D * 2;

  const int nx = M * D, nw = D * D;
  convert_f32_bf16<<<(nx + 2047) / 2048, 256, 0, stream>>>(x,  xb,  nx);
  convert_f32_bf16<<<(nw + 2047) / 2048, 256, 0, stream>>>(Wq, Wqb, nw);
  convert_f32_bf16<<<(nw + 2047) / 2048, 256, 0, stream>>>(Wk, Wkb, nw);
  convert_f32_bf16<<<(nw + 2047) / 2048, 256, 0, stream>>>(Wv, Wvb, nw);
  convert_f32_bf16<<<(nw + 2047) / 2048, 256, 0, stream>>>(Wo, Wob, nw);

  dim3 g1(D / 128, M / 128), blk(256);
  gemm_bf16_wmma<true,  false><<<g1, blk, 0, stream>>>(xb, Wqb, bq,      Qb, M, D, D);
  gemm_bf16_wmma<false, false><<<g1, blk, 0, stream>>>(xb, Wkb, nullptr, Kb, M, D, D);
  gemm_bf16_wmma<true,  false><<<g1, blk, 0, stream>>>(xb, Wvb, bv,      Vb, M, D, D);

  dim3 ga(2048 / 128, 16, 2);
  flash_attn_relbias<<<ga, blk, 0, stream>>>(Qb, Kb, Vb, pos_bias, AOb);

  gemm_bf16_wmma<true, true><<<g1, blk, 0, stream>>>(AOb, Wob, bo, d_out, M, D, D);
}
